// Attention_20495583936791
// MI455X (gfx1250) — matrix operand order repaired and checked by reading, not compiled, not run on hardware
//
#include <hip/hip_runtime.h>
#include <hip/hip_bf16.h>

// ---------------------------------------------------------------------------
// MHA block for MI455X (gfx1250): bf16 WMMA GEMMs + flash-attention.
//   - v_wmma_f32_16x16x32_bf16 for all matmuls (fp32 accumulate)
//   - global_load_async_to_lds_b128 (ASYNCcnt) for tile staging
//   - double-buffered LDS in the GEMM for load/compute overlap
// ---------------------------------------------------------------------------

typedef __bf16 v16bf __attribute__((ext_vector_type(16)));
typedef __bf16 v8bf  __attribute__((ext_vector_type(8)));
typedef float  v8f   __attribute__((ext_vector_type(8)));

#define BATCH   4
#define NSEQ    2048
#define DIM     1024
#define HEADS   16
#define HEADDIM 64
#define SCALE   0.125f    // 1/sqrt(64)

__device__ __forceinline__ v8f wmma_bf16(v16bf a, v16bf b, v8f c) {
    return __builtin_amdgcn_wmma_f32_16x16x32_bf16(
        /*neg_a=*/false, a, /*neg_b=*/false, b,
        /*c_mod=*/(short)0, c, /*reuse_a=*/false, /*reuse_b=*/false);
}

// Build a 16-element bf16 fragment from two 16-byte chunks (ds_load_b128 x2).
__device__ __forceinline__ v16bf make16(const __bf16* p0, const __bf16* p1) {
    v8bf lo = *(const v8bf*)p0;
    v8bf hi = *(const v8bf*)p1;
    v16bf r;
#pragma unroll
    for (int i = 0; i < 8; ++i) { r[i] = lo[i]; r[i + 8] = hi[i]; }
    return r;
}

// Async DMA: 16 bytes global -> LDS, tracked by ASYNCcnt.
// lds_addr is the wave-relative LDS byte offset (generic shared ptr truncated:
// low 32 bits of the address).
__device__ __forceinline__ void async_ld_b128(unsigned lds_addr, const void* g) {
    asm volatile("global_load_async_to_lds_b128 %0, %1, off"
                 :: "v"(lds_addr),
                    "v"((unsigned long long)(size_t)g)
                 : "memory");
}
__device__ __forceinline__ void wait_async0() {
    asm volatile("s_wait_asynccnt 0x0" ::: "memory");
}

// ---------------------------------------------------------------------------
// Precision-conversion helpers
// ---------------------------------------------------------------------------
__global__ void cvt_f32_to_bf16(const float* __restrict__ src,
                                __bf16* __restrict__ dst, int n) {
    int i = blockIdx.x * 256 + threadIdx.x;
    if (i < n) dst[i] = (__bf16)src[i];
}

// src [R][C] fp32 row-major  ->  dst [C][R] bf16 (i.e. transposed)
__global__ void cvt_transpose_bf16(const float* __restrict__ src,
                                   __bf16* __restrict__ dst, int R, int C) {
    int i = blockIdx.x * 256 + threadIdx.x;
    if (i < R * C) {
        int r = i / C, c = i % C;
        dst[(size_t)c * R + r] = (__bf16)src[(size_t)r * C + c];
    }
}

// ---------------------------------------------------------------------------
// WMMA GEMM:  out = A[M,K] * BT[N,K]^T + bias
//   mode 0: scatter bf16 into Q/K/V [B,H,N,64]   (QKV projection)
//   mode 1: fp32 out[M,N]                        (output projection)
// Block: 256 threads (8 waves).  Tile: 128(M) x 64(N), K-step 64.
// Waves 4(M) x 2(N); each wave owns 32x32 = 2x2 WMMA accumulators, 8 WMMA/step.
// LDS double-buffered; next slab DMA'd (async-to-LDS) during compute.
// ---------------------------------------------------------------------------
__global__ __launch_bounds__(256) void gemm_bf16_wmma(
    const __bf16* __restrict__ A, const __bf16* __restrict__ BT,
    const float* __restrict__ bias, int M, int N, int K, int mode,
    __bf16* __restrict__ qout, __bf16* __restrict__ kout,
    __bf16* __restrict__ vout, float* __restrict__ fout)
{
    __shared__ __align__(16) __bf16 As[2][128][72];  // 64-wide slab +8 pad
    __shared__ __align__(16) __bf16 Bs[2][64][72];

    const int tid  = threadIdx.x;
    const int wave = tid >> 5, lane = tid & 31;
    const int half = lane >> 4, l16 = lane & 15;
    const int wM = wave & 3, wN = wave >> 2;
    const int m0 = blockIdx.y * 128, n0 = blockIdx.x * 64;

    // Stage one 64-deep K slab into buffer `bsel` (async DMA, no VGPR bounce).
    auto stage = [&](int bsel, int k0) {
        {   // A: 128 rows x 64 bf16; 2 threads/row, 64B each = 4x b128
            int r = tid >> 1, h2 = tid & 1;
            const __bf16* g = A + (size_t)(m0 + r) * K + k0 + h2 * 32;
            unsigned l = (unsigned)(size_t)&As[bsel][r][h2 * 32];
#pragma unroll
            for (int c = 0; c < 4; ++c)
                async_ld_b128(l + c * 16, g + c * 8);
        }
        {   // B: 64 rows x 64 bf16; 4 threads/row, 32B each = 2x b128
            int r = tid >> 2, q4 = tid & 3;
            const __bf16* g = BT + (size_t)(n0 + r) * K + k0 + q4 * 16;
            unsigned l = (unsigned)(size_t)&Bs[bsel][r][q4 * 16];
            async_ld_b128(l, g);
            async_ld_b128(l + 16, g + 8);
        }
    };

    v8f acc[2][2] = {};

    int buf = 0;
    stage(buf, 0);
    wait_async0();
    __syncthreads();

    for (int k0 = 0; k0 < K; k0 += 64) {
        int nxt = buf ^ 1;
        if (k0 + 64 < K) stage(nxt, k0 + 64);    // DMA next slab during compute

        v16bf af[2][2], bfr[2][2];
#pragma unroll
        for (int mt = 0; mt < 2; ++mt) {
            int r = wM * 32 + mt * 16 + l16;                 // lane%16 = M
#pragma unroll
            for (int kc = 0; kc < 2; ++kc)
                af[mt][kc] = make16(&As[buf][r][kc * 32 + half * 8],
                                    &As[buf][r][kc * 32 + 16 + half * 8]);
        }
#pragma unroll
        for (int nt = 0; nt < 2; ++nt) {
            int r = wN * 32 + nt * 16 + l16;                 // lane%16 = N
#pragma unroll
            for (int kc = 0; kc < 2; ++kc)
                bfr[nt][kc] = make16(&Bs[buf][r][kc * 32 + half * 8],
                                     &Bs[buf][r][kc * 32 + 16 + half * 8]);
        }
#pragma unroll
        for (int mt = 0; mt < 2; ++mt)
#pragma unroll
            for (int nt = 0; nt < 2; ++nt)
#pragma unroll
                for (int kc = 0; kc < 2; ++kc)
                    acc[mt][nt] = wmma_bf16(af[mt][kc], bfr[nt][kc], acc[mt][nt]);

        wait_async0();        // my wave's DMA for the next slab has landed
        __syncthreads();      // everyone done reading `buf` / writing `nxt`
        buf = nxt;
    }

    // ---- epilogue ----
#pragma unroll
    for (int mt = 0; mt < 2; ++mt) {
#pragma unroll
        for (int nt = 0; nt < 2; ++nt) {
            int col = n0 + wN * 32 + nt * 16 + l16;
            float bs = bias[col];
            if (mode == 0) {
                int qkvi = col >> 10;              // which of q/k/v
                int hh   = (col & 1023) >> 6;      // head
                int d    = col & 63;               // within head
                __bf16* dst = (qkvi == 0) ? qout : (qkvi == 1) ? kout : vout;
#pragma unroll
                for (int j = 0; j < 8; ++j) {
                    int row = m0 + wM * 32 + mt * 16 + j + 8 * half;
                    int bb = row >> 11;            // batch
                    int nn = row & 2047;           // seq pos
                    size_t idx = ((((size_t)bb * HEADS + hh) * NSEQ) + nn) * HEADDIM + d;
                    dst[idx] = (__bf16)(acc[mt][nt][j] + bs);
                }
            } else {
#pragma unroll
                for (int j = 0; j < 8; ++j) {
                    int row = m0 + wM * 32 + mt * 16 + j + 8 * half;
                    fout[(size_t)row * N + col] = acc[mt][nt][j] + bs;
                }
            }
        }
    }
}

// ---------------------------------------------------------------------------
// Flash attention: one workgroup (4 waves) handles 64 query rows of one (b,h).
// Each wave owns 16 rows. Keys processed 32 at a time.
// K tile staged by async-to-LDS DMA; V transposed through VGPRs.
// ---------------------------------------------------------------------------
__global__ __launch_bounds__(128) void attn_flash_wmma(
    const __bf16* __restrict__ Q, const __bf16* __restrict__ K,
    const __bf16* __restrict__ V, __bf16* __restrict__ ctx)
{
    __shared__ __align__(16) __bf16 Kt[32][72];      // keys x 64 (+8 pad)
    __shared__ __align__(16) __bf16 Vt[64][40];      // V^T: [d][key] (+8 pad)
    __shared__ __align__(16) __bf16 Pt[4][16][40];   // per-wave P relayout

    const int tid  = threadIdx.x;
    const int wave = tid >> 5, lane = tid & 31;
    const int half = lane >> 4, l16 = lane & 15;
    const int mtile = blockIdx.x, h = blockIdx.y, b = blockIdx.z;
    const int bh = b * HEADS + h;

    const __bf16* Qb = Q + (size_t)bh * NSEQ * HEADDIM;
    const __bf16* Kb = K + (size_t)bh * NSEQ * HEADDIM;
    const __bf16* Vb = V + (size_t)bh * NSEQ * HEADDIM;

    // This wave's Q A-fragments, loaded once (reused for every key tile).
    v16bf aQ[2];
    {
        int row = mtile * 64 + wave * 16 + l16;      // lane%16 = M
        const __bf16* qr = Qb + (size_t)row * HEADDIM;
#pragma unroll
        for (int kc = 0; kc < 2; ++kc)
            aQ[kc] = make16(qr + kc * 32 + half * 8, qr + kc * 32 + 16 + half * 8);
    }

    v8f Oc[4] = {};                 // 16x64 fp32 accumulator (4 x 16x16)
    float mrow[8], lrow[8];
#pragma unroll
    for (int j = 0; j < 8; ++j) { mrow[j] = -3.0e38f; lrow[j] = 0.0f; }

    for (int j0 = 0; j0 < NSEQ; j0 += 32) {
        // K tile: async DMA straight to LDS (overlaps the V transpose below).
        {
            int r = tid >> 2, ch = tid & 3;          // 32 rows x 4 chunks(32B)
            const __bf16* g = Kb + (size_t)(j0 + r) * HEADDIM + ch * 16;
            unsigned l = (unsigned)(size_t)&Kt[r][ch * 16];
            async_ld_b128(l, g);
            async_ld_b128(l + 16, g + 8);
        }
        // V^T tile: Vt[d][key] = V[j0+key][d]  (VALU transpose).
        {
            int r = tid >> 2, ch = tid & 3;
            const __bf16* g = Vb + (size_t)(j0 + r) * HEADDIM + ch * 16;
            v8bf x0 = *(const v8bf*)g;
            v8bf x1 = *(const v8bf*)(g + 8);
#pragma unroll
            for (int i = 0; i < 8; ++i) {
                Vt[ch * 16 + i][r]     = x0[i];
                Vt[ch * 16 + 8 + i][r] = x1[i];
            }
        }
        wait_async0();
        __syncthreads();

        // S = Q * K^T for 32 keys -> two 16x16 tiles, each over 2 K-chunks.
        v8f S0 = {}, S1 = {};
#pragma unroll
        for (int kc = 0; kc < 2; ++kc) {
            v16bf bk0 = make16(&Kt[l16][kc * 32 + half * 8],
                               &Kt[l16][kc * 32 + 16 + half * 8]);
            v16bf bk1 = make16(&Kt[16 + l16][kc * 32 + half * 8],
                               &Kt[16 + l16][kc * 32 + 16 + half * 8]);
            S0 = wmma_bf16(aQ[kc], bk0, S0);
            S1 = wmma_bf16(aQ[kc], bk1, S1);
        }

        // Online softmax across the 32 new columns.
        float p0[8], p1[8], alpha[8];
#pragma unroll
        for (int j = 0; j < 8; ++j) {
            float s0 = S0[j] * SCALE, s1 = S1[j] * SCALE;
            float rm = fmaxf(s0, s1);
            rm = fmaxf(rm, __shfl_xor(rm, 1));
            rm = fmaxf(rm, __shfl_xor(rm, 2));
            rm = fmaxf(rm, __shfl_xor(rm, 4));
            rm = fmaxf(rm, __shfl_xor(rm, 8));
            float mnew = fmaxf(mrow[j], rm);
            alpha[j] = __expf(mrow[j] - mnew);
            mrow[j]  = mnew;
            p0[j] = __expf(s0 - mnew);
            p1[j] = __expf(s1 - mnew);
            float rs = p0[j] + p1[j];
            rs += __shfl_xor(rs, 1);
            rs += __shfl_xor(rs, 2);
            rs += __shfl_xor(rs, 4);
            rs += __shfl_xor(rs, 8);
            lrow[j] = lrow[j] * alpha[j] + rs;
        }
#pragma unroll
        for (int c = 0; c < 4; ++c)
#pragma unroll
            for (int j = 0; j < 8; ++j)
                Oc[c][j] *= alpha[j];

        // Relayout P (C-layout) -> A-fragment layout via per-wave LDS buffer.
#pragma unroll
        for (int j = 0; j < 8; ++j) {
            int m = j + 8 * half;
            Pt[wave][m][l16]      = (__bf16)p0[j];
            Pt[wave][m][16 + l16] = (__bf16)p1[j];
        }
        // Intra-wave LDS store->load (LDS is in-order per wave).
        asm volatile("s_wait_dscnt 0x0" ::: "memory");
        v16bf aP = make16(&Pt[wave][l16][half * 8], &Pt[wave][l16][16 + half * 8]);

    // O += P * V   (B-fragment column d from Vt row, table-order keys)
#pragma unroll
        for (int c = 0; c < 4; ++c) {
            v16bf bV = make16(&Vt[c * 16 + l16][half * 8],
                              &Vt[c * 16 + l16][16 + half * 8]);
            Oc[c] = wmma_bf16(aP, bV, Oc[c]);
        }
        __syncthreads();   // before next iteration overwrites Kt/Vt
    }

    // Normalize and write context in [B, N, H*64] layout (proj-GEMM A input).
#pragma unroll
    for (int j = 0; j < 8; ++j) {
        float inv = 1.0f / lrow[j];
        int nrow = mtile * 64 + wave * 16 + j + 8 * half;
        size_t base = ((size_t)b * NSEQ + nrow) * DIM + h * HEADDIM;
#pragma unroll
        for (int c = 0; c < 4; ++c)
            ctx[base + c * 16 + l16] = (__bf16)(Oc[c][j] * inv);
    }
}

// ---------------------------------------------------------------------------
// Launch: convert -> QKV GEMM -> flash attention -> proj GEMM
// ---------------------------------------------------------------------------
extern "C" void kernel_launch(void* const* d_in, const int* in_sizes, int n_in,
                              void* d_out, int out_size, void* d_ws, size_t ws_size,
                              hipStream_t stream) {
    const float* x      = (const float*)d_in[0];
    const float* w_qkv  = (const float*)d_in[1];
    const float* b_qkv  = (const float*)d_in[2];
    const float* w_proj = (const float*)d_in[3];
    const float* b_proj = (const float*)d_in[4];
    float* out = (float*)d_out;

    const size_t M = (size_t)BATCH * NSEQ;          // 8192
    char* ws = (char*)d_ws;
    size_t off = 0;
    __bf16* xb     = (__bf16*)(ws + off); off += M * DIM * 2;
    __bf16* wqkvT  = (__bf16*)(ws + off); off += (size_t)3 * DIM * DIM * 2;
    __bf16* wprojT = (__bf16*)(ws + off); off += (size_t)DIM * DIM * 2;
    __bf16* qb     = (__bf16*)(ws + off); off += M * DIM * 2;
    __bf16* kb     = (__bf16*)(ws + off); off += M * DIM * 2;
    __bf16* vb     = (__bf16*)(ws + off); off += M * DIM * 2;
    __bf16* ctxb   = (__bf16*)(ws + off); off += M * DIM * 2;

    int nx = (int)(M * DIM);
    cvt_f32_to_bf16<<<(nx + 255) / 256, 256, 0, stream>>>(x, xb, nx);
    cvt_transpose_bf16<<<(DIM * 3 * DIM + 255) / 256, 256, 0, stream>>>(
        w_qkv, wqkvT, DIM, 3 * DIM);
    cvt_transpose_bf16<<<(DIM * DIM + 255) / 256, 256, 0, stream>>>(
        w_proj, wprojT, DIM, DIM);

    // QKV projection: [8192,1024] x [1024,3072] -> scatter bf16 Q/K/V
    gemm_bf16_wmma<<<dim3(3 * DIM / 64, M / 128), 256, 0, stream>>>(
        xb, wqkvT, b_qkv, (int)M, 3 * DIM, DIM, /*mode=*/0,
        qb, kb, vb, nullptr);

    // Flash attention: grid (N/64, H, B)
    attn_flash_wmma<<<dim3(NSEQ / 64, HEADS, BATCH), 128, 0, stream>>>(
        qb, kb, vb, ctxb);

    // Output projection: [8192,1024] x [1024,1024] -> fp32 out (+bias)
    gemm_bf16_wmma<<<dim3(DIM / 64, M / 128), 256, 0, stream>>>(
        ctxb, wprojT, b_proj, (int)M, DIM, DIM, /*mode=*/1,
        nullptr, nullptr, nullptr, out);
}
